// Lfm2ShortConv_62294205661476
// MI455X (gfx1250) — matrix operand order repaired and checked by reading, not compiled, not run on hardware
//
#include <hip/hip_runtime.h>
#include <cstdint>
#include <cstddef>

// ---------------------------------------------------------------------------
// LFM2 short-conv block for MI455X (gfx1250, wave32, WMMA bf16)
//   GEMM1 (bf16 WMMA, async-LDS staged) -> BCx[8192,6144] bf16
//   depthwise causal conv (L=3) + gates -> y packed as WMMA A-fragments
//   GEMM2 (bf16 WMMA, async-LDS staged) -> out[8192,2048] fp32
// ---------------------------------------------------------------------------

typedef __attribute__((ext_vector_type(16))) __bf16 v16bf;
typedef __attribute__((ext_vector_type(8)))  float  v8f;

#define HID    2048
#define SEQ    4096
#define NBATCH 2
#define MTOT   (NBATCH * SEQ)   // 8192 rows
#define NP1    (3 * HID)        // 6144 (in_proj out features)
#define KT_H   (HID / 32)       // 64 k-tiles when K == HID

static __device__ __forceinline__ unsigned short f2bf(float f) {
  unsigned int u = __float_as_uint(f);
  u += 0x7FFFu + ((u >> 16) & 1u);       // round-to-nearest-even
  return (unsigned short)(u >> 16);
}
static __device__ __forceinline__ float bf2f(unsigned short h) {
  return __uint_as_float(((unsigned int)h) << 16);
}

// ---------------------------------------------------------------------------
// A-fragment packing (16x32 bf16 tile, ISA 7.12.2 16-bit A layout):
//   lane 0-15  : row M=lane,     elems 0-7 -> K 0-7,  elems 8-15 -> K 16-23
//   lane 16-31 : row M=lane-16,  elems 0-7 -> K 8-15, elems 8-15 -> K 24-31
// Fragment f = mt*ktiles + kt; storage: frag*512 + lane*16 + e (contig 32B/lane)
// ---------------------------------------------------------------------------
__global__ __launch_bounds__(256)
void pack_a_kernel(const float* __restrict__ src, unsigned short* __restrict__ dst,
                   int Mrows, int K) {
  int gid    = blockIdx.x * blockDim.x + threadIdx.x;
  int ktiles = K >> 5;
  int nfrag  = (Mrows >> 4) * ktiles;
  int frag   = gid >> 5;
  if (frag >= nfrag) return;
  int lane = gid & 31;
  int mt   = frag / ktiles;
  int kt   = frag - mt * ktiles;
  int half = lane >> 4;
  int m    = (mt << 4) + (lane & 15);
  const float* row = src + (size_t)m * K + ((size_t)kt << 5);
  unsigned short* out = dst + ((size_t)frag << 9) + (lane << 4);
#pragma unroll
  for (int e = 0; e < 8; ++e) out[e]     = f2bf(row[e + half * 8]);
#pragma unroll
  for (int e = 0; e < 8; ++e) out[8 + e] = f2bf(row[16 + e + half * 8]);
}

// ---------------------------------------------------------------------------
// B-fragment packing (32x16 bf16 tile): B[k,n] = W[n,k] (W is [N,K] row-major)
//   lane 0-15  : col N=lane,     elems 0-7 -> K 0-7,  elems 8-15 -> K 16-23
//   lane 16-31 : col N=lane-16,  elems 0-7 -> K 8-15, elems 8-15 -> K 24-31
// ---------------------------------------------------------------------------
__global__ __launch_bounds__(256)
void pack_b_kernel(const float* __restrict__ W, unsigned short* __restrict__ dst,
                   int N, int K) {
  int gid    = blockIdx.x * blockDim.x + threadIdx.x;
  int ktiles = K >> 5;
  int nfrag  = (N >> 4) * ktiles;
  int frag   = gid >> 5;
  if (frag >= nfrag) return;
  int lane  = gid & 31;
  int nt    = frag / ktiles;
  int kt    = frag - nt * ktiles;
  int n     = (nt << 4) + (lane & 15);
  int kbase = (kt << 5) + ((lane >> 4) << 3);
  const float* row = W + (size_t)n * K + kbase;
  unsigned short* out = dst + ((size_t)frag << 9) + (lane << 4);
#pragma unroll
  for (int e = 0; e < 8; ++e) { out[e] = f2bf(row[e]); out[8 + e] = f2bf(row[16 + e]); }
}

// ---------------------------------------------------------------------------
// WMMA GEMM with async global->LDS double buffering.
// 256 threads = 8 waves; block tile 128x128; wave tile 64x32 (4x2 WMMA tiles).
// Per K-step the block stages 8 A-frags + 8 B-frags (16 KB) in LDS; wave w
// DMA-copies A-slot w and B-slot w (4 x global_load_async_to_lds_b128).
// ---------------------------------------------------------------------------
template <bool BF16_OUT>
__global__ __launch_bounds__(256)
void gemm_wmma_kernel(const unsigned short* __restrict__ Apack,
                      const unsigned short* __restrict__ Bpack,
                      void* __restrict__ Out, int Ncols, int K) {
  __shared__ unsigned short smem[2 * 16 * 512];   // 2 buffers x 16 frags x 1KB

  const int ktiles = K >> 5;
  const int tid  = threadIdx.x;
  const int lane = tid & 31;
  const int wave = tid >> 5;
  const int wm   = wave & 1;
  const int wn   = wave >> 1;
  const int mt0  = blockIdx.x * 8 + wm * 4;   // 16-row tile index
  const int nt0  = blockIdx.y * 8 + wn * 2;   // 16-col tile index

  // this wave's DMA sources: A-frag row (bm*8+wave), B-frag col (bn*8+wave)
  const char* gA = (const char*)(Apack + (((size_t)(blockIdx.x * 8 + wave) * ktiles) << 9));
  const char* gB = (const char*)(Bpack + (((size_t)(blockIdx.y * 8 + wave) * ktiles) << 9));
  const unsigned int lbase = (unsigned int)(size_t)(&smem[0]); // LDS byte offset

  v8f c[4][2];
#pragma unroll
  for (int i = 0; i < 4; ++i)
#pragma unroll
    for (int j = 0; j < 2; ++j) {
      v8f z = {};
      c[i][j] = z;
    }

  auto issue_copies = [&](int kt, int buf) {
    const char* ga = gA + ((size_t)kt << 10) + lane * 16;
    const char* gb = gB + ((size_t)kt << 10) + lane * 16;
    unsigned int la = lbase + (unsigned int)(buf * 16384 + wave * 1024 + lane * 16);
    unsigned int lb = la + 8192;   // B slots live in second 8KB of the buffer
    asm volatile("global_load_async_to_lds_b128 %0, %1, off"
                 :: "v"(la), "v"(ga) : "memory");
    asm volatile("global_load_async_to_lds_b128 %0, %1, off"
                 :: "v"(la + 512u), "v"(ga + 512) : "memory");
    asm volatile("global_load_async_to_lds_b128 %0, %1, off"
                 :: "v"(lb), "v"(gb) : "memory");
    asm volatile("global_load_async_to_lds_b128 %0, %1, off"
                 :: "v"(lb + 512u), "v"(gb + 512) : "memory");
    // keep the L2 warm a few steps ahead (-> global_prefetch_b8)
    __builtin_prefetch(ga + 8 * 1024, 0, 1);
    __builtin_prefetch(gb + 8 * 1024, 0, 1);
  };

  issue_copies(0, 0);                               // prologue: fill buffer 0

  for (int kt = 0; kt < ktiles; ++kt) {
    const int cur = kt & 1;
    if (kt + 1 < ktiles) {
      issue_copies(kt + 1, cur ^ 1);                // overlap DMA with compute
      asm volatile("s_wait_asynccnt 0x4" ::: "memory");  // buffer `cur` landed
    } else {
      asm volatile("s_wait_asynccnt 0x0" ::: "memory");
    }
    __syncthreads();                                // all waves' copies visible

    const unsigned short* sb = smem + cur * 16 * 512;
    v16bf a[4], b[2];
#pragma unroll
    for (int i = 0; i < 4; ++i)
      a[i] = *(const v16bf*)(sb + (wm * 4 + i) * 512 + lane * 16);
#pragma unroll
    for (int j = 0; j < 2; ++j)
      b[j] = *(const v16bf*)(sb + (8 + wn * 2 + j) * 512 + lane * 16);

#pragma unroll
    for (int i = 0; i < 4; ++i)
#pragma unroll
      for (int j = 0; j < 2; ++j)
        c[i][j] = __builtin_amdgcn_wmma_f32_16x16x32_bf16(
            /*neg_a=*/false, a[i], /*neg_b=*/false, b[j],
            /*c_mod=*/(short)0, c[i][j], /*reuse_a=*/false, /*reuse_b=*/false);

    __syncthreads();   // reads done before this buffer is overwritten next step
  }

  // C layout (ISA 7.12.2): VGPR r of lane l -> row (r + (l>>4)*8), col (l&15)
  const int half = lane >> 4;
  const int ncol = lane & 15;
#pragma unroll
  for (int i = 0; i < 4; ++i) {
    const int mbase = (mt0 + i) * 16 + half * 8;
#pragma unroll
    for (int j = 0; j < 2; ++j) {
      const int n = (nt0 + j) * 16 + ncol;
#pragma unroll
      for (int r = 0; r < 8; ++r) {
        const size_t off = (size_t)(mbase + r) * (size_t)Ncols + n;
        if (BF16_OUT)
          ((unsigned short*)Out)[off] = f2bf(c[i][j][r]);
        else
          ((float*)Out)[off] = c[i][j][r];
      }
    }
  }
}

// ---------------------------------------------------------------------------
// Depthwise causal conv (L=3) + gates, writing y directly as packed
// A-fragments for GEMM2.  BCx row-major [8192, 6144] bf16:
//   Bg = BCx[:, 0:H], Cg = BCx[:, H:2H], x = BCx[:, 2H:3H]
//   Bx(s) = Bg(s)*x(s);  conv(s) = w0*Bx(s-2)+w1*Bx(s-1)+w2*Bx(s)
//   y(s)  = Cg(s)*conv(s)
// ---------------------------------------------------------------------------
__global__ __launch_bounds__(256)
void conv_gate_pack_kernel(const unsigned short* __restrict__ bcx,
                           const float* __restrict__ wconv,
                           unsigned short* __restrict__ ypack) {
  int gid  = blockIdx.x * blockDim.x + threadIdx.x;
  int frag = gid >> 5;
  const int ktiles = KT_H;                 // 64 (K = HID)
  if (frag >= (MTOT >> 4) * ktiles) return;
  int lane = gid & 31;
  int mt   = frag / ktiles;
  int kt   = frag - mt * ktiles;
  int half = lane >> 4;
  int m    = (mt << 4) + (lane & 15);
  int s    = m & (SEQ - 1);

  const size_t row0 = (size_t)m * NP1;
  const size_t row1 = row0 - NP1;          // s-1 (valid iff s>=1)
  const size_t row2 = row0 - 2 * NP1;      // s-2 (valid iff s>=2)
  unsigned short* out = ypack + ((size_t)frag << 9) + (lane << 4);

#pragma unroll
  for (int e = 0; e < 16; ++e) {
    int klocal = (e < 8) ? (e + half * 8) : (16 + (e - 8) + half * 8);
    int h = (kt << 5) + klocal;
    float w0 = wconv[h * 3 + 0];
    float w1 = wconv[h * 3 + 1];
    float w2 = wconv[h * 3 + 2];

    float bx0 = bf2f(bcx[row0 + h]) * bf2f(bcx[row0 + 2 * HID + h]);
    float acc = w2 * bx0;
    if (s >= 1) acc += w1 * (bf2f(bcx[row1 + h]) * bf2f(bcx[row1 + 2 * HID + h]));
    if (s >= 2) acc += w0 * (bf2f(bcx[row2 + h]) * bf2f(bcx[row2 + 2 * HID + h]));

    float cg = bf2f(bcx[row0 + HID + h]);
    out[e] = f2bf(cg * acc);
  }
}

// ---------------------------------------------------------------------------
extern "C" void kernel_launch(void* const* d_in, const int* in_sizes, int n_in,
                              void* d_out, int out_size, void* d_ws, size_t ws_size,
                              hipStream_t stream) {
  const float* hs     = (const float*)d_in[0];  // [2,4096,2048]
  const float* W_in   = (const float*)d_in[1];  // [6144,2048]
  const float* W_conv = (const float*)d_in[2];  // [2048,1,3]
  const float* W_out  = (const float*)d_in[3];  // [2048,2048]
  float* out = (float*)d_out;                   // [2,4096,2048] fp32

  // carve workspace (bytes), 256-aligned chunks
  char* ws = (char*)d_ws;
  size_t off = 0;
  auto carve = [&](size_t bytes) {
    char* p = ws + off;
    off += (bytes + 255) & ~(size_t)255;
    return p;
  };
  unsigned short* xpack = (unsigned short*)carve((size_t)MTOT * HID * 2);  // 32 MB
  unsigned short* winp  = (unsigned short*)carve((size_t)NP1  * HID * 2);  // 24 MB
  unsigned short* wop   = (unsigned short*)carve((size_t)HID  * HID * 2);  //  8 MB
  unsigned short* bcx   = (unsigned short*)carve((size_t)MTOT * NP1 * 2);  // 96 MB
  unsigned short* ypack = (unsigned short*)carve((size_t)MTOT * HID * 2);  // 32 MB

  const int threads = 256;

  // 1) pack activations X -> bf16 A-fragments
  {
    int nfrag = (MTOT / 16) * (HID / 32);                 // 32768
    pack_a_kernel<<<(nfrag * 32 + threads - 1) / threads, threads, 0, stream>>>(
        hs, xpack, MTOT, HID);
  }
  // 2) pack W_in / W_out -> bf16 B-fragments
  {
    int nfrag = (NP1 / 16) * (HID / 32);                  // 24576
    pack_b_kernel<<<(nfrag * 32 + threads - 1) / threads, threads, 0, stream>>>(
        W_in, winp, NP1, HID);
  }
  {
    int nfrag = (HID / 16) * (HID / 32);                  // 8192
    pack_b_kernel<<<(nfrag * 32 + threads - 1) / threads, threads, 0, stream>>>(
        W_out, wop, HID, HID);
  }
  // 3) GEMM1: BCx = X * W_in^T  (bf16 out)
  {
    dim3 grid(MTOT / 128, NP1 / 128);                     // (64, 48)
    gemm_wmma_kernel<true><<<grid, threads, 0, stream>>>(xpack, winp, bcx, NP1, HID);
  }
  // 4) conv + gates -> y packed as A-fragments
  {
    int nfrag = (MTOT / 16) * (HID / 32);                 // 32768
    conv_gate_pack_kernel<<<(nfrag * 32 + threads - 1) / threads, threads, 0, stream>>>(
        bcx, W_conv, ypack);
  }
  // 5) GEMM2: out = y * W_out^T  (fp32 out)
  {
    dim3 grid(MTOT / 128, HID / 128);                     // (64, 16)
    gemm_wmma_kernel<false><<<grid, threads, 0, stream>>>(ypack, wop, out, HID, HID);
  }
}
